// AttentionWithCache_69939247448559
// MI455X (gfx1250) — compile-verified, matrix operand order checked by reading, not run on hardware
//
#include <hip/hip_runtime.h>

typedef __attribute__((ext_vector_type(16))) _Float16 v16h;
typedef __attribute__((ext_vector_type(8)))  float    v8f;

#define B_      16
#define S_      4
#define D_      4096
#define H_      32
#define DH_     128
#define CACHED_ 4096
#define ROWS_   (B_*S_)        // 64
#define JTOT_   (CACHED_+S_)   // 4100 keys per (b,h)
#define NCHUNK_ 129            // ceil(4100/32)
#define JPAD_   (NCHUNK_*32)   // 4128, padded score row stride
#define CSPLIT_ 16             // j-range split for ctx kernel
#define VSTRIDE_ 132           // padded LDS V-tile row stride (floats)

#if defined(__HIP_DEVICE_COMPILE__) && \
    __has_builtin(__builtin_amdgcn_global_load_async_to_lds_b128)
#define USE_ASYNC_LDS 1
#else
#define USE_ASYNC_LDS 0
#endif

// Exact parameter types of the async-copy builtin on this toolchain
// (from the round-2 diagnostic): non-const GCC vector int4, AS1 src / AS3 dst.
typedef int vi4_ __attribute__((vector_size(4 * sizeof(int))));
typedef __attribute__((address_space(1))) vi4_* g_vi4_ptr;
typedef __attribute__((address_space(3))) vi4_* l_vi4_ptr;

static __device__ __forceinline__ void wait_async0() {
#if USE_ASYNC_LDS
#if __has_builtin(__builtin_amdgcn_s_wait_asynccnt)
  __builtin_amdgcn_s_wait_asynccnt(0);
#else
  asm volatile("s_wait_asynccnt 0" ::: "memory");
#endif
#endif
}

// 16-byte global -> LDS copy: async DMA path on gfx1250 (ASYNCcnt tracked),
// synchronous fallback otherwise.
static __device__ __forceinline__ void copy16_g2lds(const float* __restrict__ g,
                                                    float* l) {
#if USE_ASYNC_LDS
  void* gv = const_cast<float*>(g);
  __builtin_amdgcn_global_load_async_to_lds_b128((g_vi4_ptr)gv, (l_vi4_ptr)l,
                                                 0, 0);
#else
  *(float4*)l = *(const float4*)g;
#endif
}

static __device__ __forceinline__ v8f zero8() {
  v8f a;
#pragma unroll
  for (int i = 0; i < 8; ++i) a[i] = 0.0f;
  return a;
}

static __device__ __forceinline__ v16h frag_zero() {
  v16h f;
#pragma unroll
  for (int i = 0; i < 16; ++i) f[i] = (_Float16)0.0f;
  return f;
}

// Build one lane's f16 fragment for v_wmma_f32_16x16x32_f16 from a row-major
// fp32 row. p must point at (row, kbase + 8*half). Element i -> K = 8*half+i,
// element i+8 -> K = 16 + 8*half + i (matches ISA 16-bit 16x32 A/B layout).
static __device__ __forceinline__ v16h frag_row(const float* __restrict__ p) {
  const float4 a0 = ((const float4*)p)[0];
  const float4 a1 = ((const float4*)p)[1];
  const float4 b0 = ((const float4*)p)[4];
  const float4 b1 = ((const float4*)p)[5];
  v16h f;
  f[0]  = (_Float16)a0.x; f[1]  = (_Float16)a0.y;
  f[2]  = (_Float16)a0.z; f[3]  = (_Float16)a0.w;
  f[4]  = (_Float16)a1.x; f[5]  = (_Float16)a1.y;
  f[6]  = (_Float16)a1.z; f[7]  = (_Float16)a1.w;
  f[8]  = (_Float16)b0.x; f[9]  = (_Float16)b0.y;
  f[10] = (_Float16)b0.z; f[11] = (_Float16)b0.w;
  f[12] = (_Float16)b1.x; f[13] = (_Float16)b1.y;
  f[14] = (_Float16)b1.z; f[15] = (_Float16)b1.w;
  return f;
}

static __device__ __forceinline__ v8f wmma16(v16h a, v16h b, v8f c) {
  return __builtin_amdgcn_wmma_f32_16x16x32_f16(false, a, false, b,
                                                (short)0, c, false, false);
}

// ---------------------------------------------------------------------------
// out[64, 4096] = A[64, 4096] @ W[4096,4096]^T + bias
// Grid: 256 blocks (one 16-wide N tile each), 128 threads = 4 waves.
// Each wave owns a 16x16 D tile; the W tile is converted to f16 in LDS once
// per block so every weight element is read from HBM exactly once.
// ---------------------------------------------------------------------------
__global__ void __launch_bounds__(128) gemm_xWT_kernel(
    const float* __restrict__ A, const float* __restrict__ W,
    const float* __restrict__ bias, float* __restrict__ out) {
  __shared__ _Float16 bt[16][40];  // [n][k] tile, padded vs bank conflicts

  const int n0   = blockIdx.x * 16;
  const int lane = threadIdx.x & 31;
  const int wave = threadIdx.x >> 5;   // rows 16*wave .. 16*wave+15
  const int half = lane >> 4;
  const int l15  = lane & 15;

  const float* arow = A + (size_t)(wave * 16 + l15) * D_;

  v8f acc = zero8();

  for (int kb = 0; kb < D_; kb += 32) {
    __syncthreads();
    {  // cooperative load of W[n0..n0+15][kb..kb+31] -> f16 LDS
      const int n  = threadIdx.x >> 3;        // 0..15
      const int k4 = (threadIdx.x & 7) * 4;   // 0..28
      const float4 w4 = *(const float4*)(W + (size_t)(n0 + n) * D_ + kb + k4);
      bt[n][k4 + 0] = (_Float16)w4.x;
      bt[n][k4 + 1] = (_Float16)w4.y;
      bt[n][k4 + 2] = (_Float16)w4.z;
      bt[n][k4 + 3] = (_Float16)w4.w;
    }
    __syncthreads();

    const v16h a = frag_row(arow + kb + half * 8);
    v16h b;
#pragma unroll
    for (int i = 0; i < 8; ++i) {
      b[i]     = bt[l15][8 * half + i];
      b[i + 8] = bt[l15][16 + 8 * half + i];
    }
    acc = wmma16(a, b, acc);
  }

#pragma unroll
  for (int r = 0; r < 8; ++r) {
    const int m = r + 8 * half;
    out[(size_t)(wave * 16 + m) * D_ + n0 + l15] = acc[r] + bias[n0 + l15];
  }
}

// ---------------------------------------------------------------------------
// scores[b,h,s,j] = q[b,h,s,:] . K[b,h,j,:] / sqrt(DH)
// One wave per 32-key chunk per (b,h). Lane = key column; each lane streams
// one K-cache row with contiguous float4 loads (read exactly once from HBM).
// ---------------------------------------------------------------------------
__global__ void __launch_bounds__(32) scores_kernel(
    const float* __restrict__ qf, const float* __restrict__ kf,
    const float* __restrict__ k_cache, float* __restrict__ sc) {
  const int bid   = blockIdx.x;
  const int chunk = bid % NCHUNK_;
  const int bh    = bid / NCHUNK_;
  const int b = bh / H_, h = bh % H_;

  const int lane = threadIdx.x;
  const int half = lane >> 4;
  const int l15  = lane & 15;
  const bool qvalid = (l15 < S_);

  const float* qrow = qf + (size_t)(b * S_ + (qvalid ? l15 : 0)) * D_ + h * DH_;
  v16h aq[4];
#pragma unroll
  for (int dc = 0; dc < 4; ++dc)
    aq[dc] = qvalid ? frag_row(qrow + dc * 32 + half * 8) : frag_zero();

  const int j0 = chunk * 32;
  const float scale = 0.08838834764831845f;  // 1/sqrt(128)

#pragma unroll
  for (int t = 0; t < 2; ++t) {
    const int j = j0 + 16 * t + l15;
    const float* krow = k_cache;
    bool kvalid = true;
    if (j < CACHED_)
      krow = k_cache + ((size_t)(b * H_ + h) * CACHED_ + j) * DH_;
    else if (j < JTOT_)
      krow = kf + (size_t)(b * S_ + (j - CACHED_)) * D_ + h * DH_;
    else
      kvalid = false;

    v8f acc = zero8();
#pragma unroll
    for (int dc = 0; dc < 4; ++dc) {
      const v16h bf = kvalid ? frag_row(krow + dc * 32 + half * 8) : frag_zero();
      acc = wmma16(aq[dc], bf, acc);
    }
    // D tile: lanes 0-15 hold M=r; only rows s=0..3 are real queries.
    if (half == 0) {
#pragma unroll
      for (int r = 0; r < 4; ++r)
        sc[((size_t)bh * S_ + r) * JPAD_ + j0 + 16 * t + l15] = acc[r] * scale;
    }
  }
}

// ---------------------------------------------------------------------------
// Row softmax with causal length 4097+s; zero-fills the padded tail so the
// ctx GEMM needs no masking. One 128-thread block per row.
// ---------------------------------------------------------------------------
__global__ void __launch_bounds__(128) softmax_kernel(float* __restrict__ sc) {
  const int row = blockIdx.x;          // (b*H + h)*S + s
  const int s   = row & (S_ - 1);
  const int len = CACHED_ + s + 1;
  float* p = sc + (size_t)row * JPAD_;

  __shared__ float redm[4], reds[4];
  const int tid = threadIdx.x, lane = tid & 31, wave = tid >> 5;

  float m = -3.0e38f;
  for (int i = tid; i < len; i += 128) m = fmaxf(m, p[i]);
#pragma unroll
  for (int off = 16; off > 0; off >>= 1) m = fmaxf(m, __shfl_xor(m, off, 32));
  if (lane == 0) redm[wave] = m;
  __syncthreads();
  m = fmaxf(fmaxf(redm[0], redm[1]), fmaxf(redm[2], redm[3]));

  float ssum = 0.0f;
  for (int i = tid; i < len; i += 128) ssum += __expf(p[i] - m);
#pragma unroll
  for (int off = 16; off > 0; off >>= 1) ssum += __shfl_xor(ssum, off, 32);
  if (lane == 0) reds[wave] = ssum;
  __syncthreads();
  ssum = reds[0] + reds[1] + reds[2] + reds[3];
  const float inv = 1.0f / ssum;

  for (int i = tid; i < len; i += 128) p[i] = __expf(p[i] - m) * inv;
  for (int i = len + tid; i < JPAD_; i += 128) p[i] = 0.0f;
}

// ---------------------------------------------------------------------------
// ctx[b,s,h*DH+d] = sum_j P[b,h,s,j] * V[b,h,j,d]
// Block = 8 waves (one 16-wide d tile each) x CSPLIT_ j-range groups per
// (b,h). Each 32-key x 128-d V tile (16 KB) is DMA'd global->LDS with
// GLOBAL_LOAD_ASYNC_TO_LDS_B128 (double-buffered, ASYNCcnt-tracked) so the
// 1.07 GB V stream is fully coalesced with no VGPR staging. The strided
// K-dimension gather happens in LDS (row stride 132 floats: the two 16-lane
// halves hit disjoint bank ranges). Partial 16x16 tiles are merged with
// global_atomic_add_f32 into the zeroed ctx accumulator.
// ---------------------------------------------------------------------------
__global__ void __launch_bounds__(256) ctx_kernel(
    const float* __restrict__ sc, const float* __restrict__ v_cache,
    const float* __restrict__ vf, float* __restrict__ ctx) {
  __shared__ float vt[2][32][VSTRIDE_];

  const int bid = blockIdx.x;
  const int g   = bid % CSPLIT_;
  const int bh  = bid / CSPLIT_;
  const int b = bh / H_, h = bh % H_;

  const int tid  = threadIdx.x;
  const int wave = tid >> 5, lane = tid & 31;
  const int half = lane >> 4, l15 = lane & 15;
  const int d0 = wave * 16;

  // Cooperative V-tile loader: thread t copies 64 contiguous bytes of row
  // (jb + t/8) starting at column (t%8)*16. 256 threads cover 32x128 floats.
  const int lr = tid >> 3;         // 0..31 tile row
  const int lc = (tid & 7) * 16;   // 0..112 tile col
  const float* vsrc_cache =
      v_cache + ((size_t)(b * H_ + h) * CACHED_) * DH_ + lc;
  const float* vsrc_new = vf + (size_t)(b * S_) * D_ + h * DH_ + lc;

  const float* prow_base =
      sc + ((size_t)bh * S_ + (l15 < S_ ? l15 : 0)) * JPAD_ + half * 8;

#define CTX_LOAD_TILE(JB, BUF)                                              \
  do {                                                                      \
    const int j_ = (JB) + lr;                                               \
    float* dst_ = &vt[(BUF)][lr][lc];                                       \
    if (j_ < CACHED_) {                                                     \
      const float* s_ = vsrc_cache + (size_t)j_ * DH_;                      \
      copy16_g2lds(s_ + 0, dst_ + 0);                                       \
      copy16_g2lds(s_ + 4, dst_ + 4);                                       \
      copy16_g2lds(s_ + 8, dst_ + 8);                                       \
      copy16_g2lds(s_ + 12, dst_ + 12);                                     \
    } else if (j_ < JTOT_) {                                                \
      const float* s_ = vsrc_new + (size_t)(j_ - CACHED_) * D_;             \
      copy16_g2lds(s_ + 0, dst_ + 0);                                       \
      copy16_g2lds(s_ + 4, dst_ + 4);                                       \
      copy16_g2lds(s_ + 8, dst_ + 8);                                       \
      copy16_g2lds(s_ + 12, dst_ + 12);                                     \
    } else {                                                                \
      const float4 z_ = {0.0f, 0.0f, 0.0f, 0.0f};                           \
      *(float4*)(dst_ + 0) = z_;                                            \
      *(float4*)(dst_ + 4) = z_;                                            \
      *(float4*)(dst_ + 8) = z_;                                            \
      *(float4*)(dst_ + 12) = z_;                                           \
    }                                                                       \
  } while (0)

  v8f acc = zero8();
  int buf = 0;
  if (g < NCHUNK_) CTX_LOAD_TILE(g * 32, 0);

  for (int c = g; c < NCHUNK_; c += CSPLIT_) {
    wait_async0();       // this thread's tile portion has landed in LDS
    __syncthreads();     // ...and everyone else's; also fences buffer reuse

    const int cn = c + CSPLIT_;
    if (cn < NCHUNK_) {  // kick off next tile while computing this one
      CTX_LOAD_TILE(cn * 32, buf ^ 1);
      __builtin_prefetch(prow_base + cn * 32, 0, 1);
    }

    const int jb = c * 32;
    const v16h a = (l15 < S_) ? frag_row(prow_base + jb) : frag_zero();

    v16h bf;  // B = V tile 32x16: lane col d = d0+l15, K = j within chunk
#pragma unroll
    for (int i = 0; i < 16; ++i) {
      const int kk = (i < 8) ? (8 * half + i) : (16 + 8 * half + (i - 8));
      bf[i] = (_Float16)vt[buf][kk][d0 + l15];
    }
    acc = wmma16(a, bf, acc);
    buf ^= 1;
  }

#pragma unroll
  for (int r = 0; r < 8; ++r) {
    const int m = r + 8 * half;
    if (m < S_)
      atomicAdd(&ctx[(size_t)(b * S_ + m) * D_ + h * DH_ + d0 + l15], acc[r]);
  }
#undef CTX_LOAD_TILE
}

__global__ void zero_kernel(float* __restrict__ p, int n) {
  const int i = blockIdx.x * blockDim.x + threadIdx.x;
  if (i < n) p[i] = 0.0f;
}

extern "C" void kernel_launch(void* const* d_in, const int* in_sizes, int n_in,
                              void* d_out, int out_size, void* d_ws,
                              size_t ws_size, hipStream_t stream) {
  (void)in_sizes; (void)n_in; (void)out_size; (void)ws_size;
  const float* x  = (const float*)d_in[0];
  const float* kc = (const float*)d_in[1];
  const float* vc = (const float*)d_in[2];
  const float* Wq = (const float*)d_in[3];
  const float* bq = (const float*)d_in[4];
  const float* Wk = (const float*)d_in[5];
  const float* bk = (const float*)d_in[6];
  const float* Wv = (const float*)d_in[7];
  const float* bv = (const float*)d_in[8];
  const float* Wo = (const float*)d_in[9];
  const float* bo = (const float*)d_in[10];
  float* out = (float*)d_out;

  float* ws  = (float*)d_ws;
  float* qf  = ws;                       // [64,4096]
  float* kf  = qf + (size_t)ROWS_ * D_;  // [64,4096] new keys
  float* vf  = kf + (size_t)ROWS_ * D_;  // [64,4096] new values
  float* ctx = vf + (size_t)ROWS_ * D_;  // [64,4096] accumulator
  float* sc  = ctx + (size_t)ROWS_ * D_; // [B*H*S, JPAD_] scores

  const int nctx = ROWS_ * D_;
  zero_kernel<<<(nctx + 255) / 256, 256, 0, stream>>>(ctx, nctx);

  gemm_xWT_kernel<<<D_ / 16, 128, 0, stream>>>(x, Wq, bq, qf);
  gemm_xWT_kernel<<<D_ / 16, 128, 0, stream>>>(x, Wk, bk, kf);
  gemm_xWT_kernel<<<D_ / 16, 128, 0, stream>>>(x, Wv, bv, vf);

  scores_kernel<<<B_ * H_ * NCHUNK_, 32, 0, stream>>>(qf, kf, kc, sc);
  softmax_kernel<<<B_ * H_ * S_, 128, 0, stream>>>(sc);
  ctx_kernel<<<B_ * H_ * CSPLIT_, 256, 0, stream>>>(sc, vc, vf, ctx);

  gemm_xWT_kernel<<<D_ / 16, 128, 0, stream>>>(ctx, Wo, bo, out);
}
